// SelfAttention_gate2_72060961292403
// MI455X (gfx1250) — matrix operand order repaired and checked by reading, not compiled, not run on hardware
//
#include <hip/hip_runtime.h>
#include <hip/hip_bf16.h>

typedef __bf16 bf16;
typedef __attribute__((ext_vector_type(16))) bf16  v16bf;
typedef __attribute__((ext_vector_type(8)))  bf16  v8bf;
typedef __attribute__((ext_vector_type(8)))  float v8f;
typedef __attribute__((ext_vector_type(4)))  float v4f;
typedef __attribute__((ext_vector_type(4)))  unsigned v4u;

#define N_NODES_C 30000
#define T_C       6
#define EMBED_C   512
#define HEADS_C   8
#define HDIM_C    64
#define MROWS_C   (N_NODES_C * T_C)   /* 180000 = 3750 * 48 */
#define NGRAPH_C  512
#define MTILE_C   48                  /* rows per GEMM block (3 x 16) */

// ---------------------------------------------------------------------------
// Kernel 1: pack Wq (pre-scaled by 1/sqrt(HEAD_DIM)=0.125) and Wk into one
// bf16 weight matrix Wc[1024][512]; row j<512 -> Wq row j, j>=512 -> Wk.
// Reference computes X @ W.T => out[j] = sum_c X[c]*W[j][c]; row-major [j][c]
// is already K-contiguous per output column (the WMMA B layout we load).
// ---------------------------------------------------------------------------
__global__ void cvt_weights(const float* __restrict__ Wq,
                            const float* __restrict__ Wk,
                            bf16* __restrict__ Wc) {
    int i = blockIdx.x * 256 + threadIdx.x;
    if (i < EMBED_C * EMBED_C) {
        Wc[i]                     = (bf16)(Wq[i] * 0.125f);
        Wc[EMBED_C * EMBED_C + i] = (bf16)(Wk[i]);
    }
}

// ---------------------------------------------------------------------------
// Kernel 2: gate table  gates[g][t] = ac[g] . Wg[t] + bg[t]   (512 x 6)
// ---------------------------------------------------------------------------
__global__ void gate_kernel(const float* __restrict__ ac,
                            const float* __restrict__ Wg,
                            const float* __restrict__ bg,
                            float* __restrict__ gates) {
    int i = blockIdx.x * 256 + threadIdx.x;
    if (i < NGRAPH_C * T_C) {
        int g = i / T_C, t = i % T_C;
        float acc = bg[t];
#pragma unroll
        for (int k = 0; k < 5; ++k) acc = fmaf(ac[g * 5 + k], Wg[t * 5 + k], acc);
        gates[i] = acc;
    }
}

// ---------------------------------------------------------------------------
// Kernel 3: projection GEMM with v_wmma_f32_16x16x32_bf16.
// Block: 512 threads (16 waves) owns a 48-row M tile (A panel staged once in
// LDS as bf16 -> X read exactly once from HBM). Wave w computes columns
// [w*64, w*64+64) for all 3 M sub-tiles, so each B fragment is reused 3x in
// registers (cuts L2->WGP weight traffic 3x vs. a 16-row tile).
// ---------------------------------------------------------------------------
__global__ __launch_bounds__(512) void proj_gemm(
        const float* __restrict__ X, const bf16* __restrict__ Wc,
        const float* __restrict__ bq, const float* __restrict__ bk,
        bf16* __restrict__ Qbf, bf16* __restrict__ Kbf) {
    __shared__ bf16 ash[MTILE_C * EMBED_C];   // 48 KB bf16 A panel

    const int tid   = threadIdx.x;
    const int Mbase = blockIdx.x * MTILE_C;

    // Cooperative, coalesced load of 48x512 f32 -> bf16 LDS
    for (int i = tid; i < MTILE_C * EMBED_C; i += 512)
        ash[i] = (bf16)X[(size_t)(Mbase + (i >> 9)) * EMBED_C + (i & 511)];
    __syncthreads();

    const int lane = tid & 31;
    const int wv   = tid >> 5;
    const int nb   = wv * 64;          // this wave's N base (0..960)
    const int m    = lane & 15;        // M row within sub-tile (A) / N col (B,C)
    const int hi   = lane >> 4;        // lane-half selector

    v8f acc[3][4] = {};                // 3 M sub-tiles x 4 N tiles of 16x16 f32

#pragma unroll 2
    for (int k0 = 0; k0 < EMBED_C; k0 += 32) {
        // A fragments: lane (m,hi) holds K = [k0+hi*16..+16)
        v16bf afrag[3];
#pragma unroll
        for (int mt = 0; mt < 3; ++mt) {
            const bf16* arow = &ash[(mt * 16 + m) * EMBED_C];
            v8bf alo = *(const v8bf*)(arow + k0 + hi * 16);
            v8bf ahi = *(const v8bf*)(arow + k0 + hi * 16 + 8);
            afrag[mt] = __builtin_shufflevector(alo, ahi,
                          0, 1, 2, 3, 4, 5, 6, 7, 8, 9, 10, 11, 12, 13, 14, 15);
        }
#pragma unroll
        for (int t = 0; t < 4; ++t) {
            // B fragment: lane n=m, K = k0 + hi*16 + [0..16): 32 contiguous bytes
            v16bf b = *(const v16bf*)(Wc + (size_t)(nb + t * 16 + m) * EMBED_C
                                         + k0 + hi * 16);
#pragma unroll
            for (int mt = 0; mt < 3; ++mt)
                acc[mt][t] = __builtin_amdgcn_wmma_f32_16x16x32_bf16(
                                 false, afrag[mt], false, b, (short)0,
                                 acc[mt][t], false, false);
        }
    }

    // Epilogue: bias (bq pre-scaled by 0.125) + bf16 store.
#pragma unroll
    for (int t = 0; t < 4; ++t) {
        int  j   = nb + t * 16 + m;                 // output column in [0,1024)
        bool isQ = (j < EMBED_C);                   // uniform per (wave, t)
        float bias = isQ ? bq[j] * 0.125f : bk[j - EMBED_C];
        bf16* op   = isQ ? Qbf : Kbf;
        int   col  = isQ ? j : j - EMBED_C;
#pragma unroll
        for (int mt = 0; mt < 3; ++mt)
#pragma unroll
            for (int v = 0; v < 8; ++v) {
                int row = Mbase + mt * 16 + v + hi * 8;   // C layout: VGPR v, half hi
                op[(size_t)row * EMBED_C + col] = (bf16)(acc[mt][t][v] + bias);
            }
    }
}

// ---------------------------------------------------------------------------
// Kernel 4: per-node gated attention. One wave per node, 2 nodes per block.
// Scores already carry the 1/sqrt(Dh) factor (folded into Wq/bq).
// All LDS access vectorized (b128) to keep the kernel HBM-bound.
// ---------------------------------------------------------------------------
__global__ __launch_bounds__(64) void attn_kernel(
        const float* __restrict__ X, const bf16* __restrict__ Qbf,
        const bf16* __restrict__ Kbf, const float* __restrict__ gates,
        const int* __restrict__ batch,
        float* __restrict__ ctx_out, float* __restrict__ attn_out) {
    __shared__ float vsh[2][T_C * EMBED_C];          // 24 KB
    __shared__ bf16  qsh[2][T_C * EMBED_C];          // 12 KB
    __shared__ bf16  ksh[2][T_C * EMBED_C];          // 12 KB
    __shared__ float ssh[2][HEADS_C * T_C * T_C];    // scores -> attention
    __shared__ float gsh[2][T_C];

    const int w    = threadIdx.x >> 5;
    const int lane = threadIdx.x & 31;
    const int n    = blockIdx.x * 2 + w;
    const size_t rowbase = (size_t)n * T_C * EMBED_C;

    // Stage V (f32, float4) and Q/K (bf16, copied as uint4)
    {
        const v4f* xs = (const v4f*)(X + rowbase);
        v4f* vd = (v4f*)vsh[w];
        for (int i = lane; i < T_C * EMBED_C / 4; i += 32) vd[i] = xs[i];
        const v4u* qs = (const v4u*)(Qbf + rowbase);
        const v4u* ks = (const v4u*)(Kbf + rowbase);
        v4u* qd = (v4u*)qsh[w];
        v4u* kd = (v4u*)ksh[w];
        for (int i = lane; i < T_C * EMBED_C / 8; i += 32) { qd[i] = qs[i]; kd[i] = ks[i]; }
    }
    if (lane < T_C) gsh[w][lane] = gates[batch[n] * T_C + lane];
    __syncthreads();

    // Scores: 8 heads x 6 x 6 dot products of length 64 (b128 LDS reads),
    // gated by key index s.
    for (int r = lane; r < HEADS_C * T_C * T_C; r += 32) {
        int h = r / (T_C * T_C);
        int t = (r / T_C) % T_C;
        int s = r % T_C;
        const v8bf* qv = (const v8bf*)&qsh[w][t * EMBED_C + h * HDIM_C];
        const v8bf* kv = (const v8bf*)&ksh[w][s * EMBED_C + h * HDIM_C];
        float acc = 0.f;
#pragma unroll
        for (int d8 = 0; d8 < HDIM_C / 8; ++d8) {
            v8bf q = qv[d8], k = kv[d8];
#pragma unroll
            for (int j = 0; j < 8; ++j)
                acc = fmaf((float)q[j], (float)k[j], acc);
        }
        ssh[w][r] = acc * gsh[w][s];
    }
    __syncthreads();

    // abs-top-4-of-6 mask + softmax; write attention, keep weights in LDS
    for (int r = lane; r < HEADS_C * T_C; r += 32) {
        int h = r / T_C, t = r % T_C;
        float sc[T_C];
#pragma unroll
        for (int s = 0; s < T_C; ++s) sc[s] = ssh[w][h * T_C * T_C + t * T_C + s];
        int i1 = 0; float m1 = fabsf(sc[0]);
#pragma unroll
        for (int s = 1; s < T_C; ++s) { float a = fabsf(sc[s]); if (a < m1) { m1 = a; i1 = s; } }
        int i2 = -1; float m2 = 3.4e38f;
#pragma unroll
        for (int s = 0; s < T_C; ++s) {
            if (s == i1) continue;
            float a = fabsf(sc[s]); if (a < m2) { m2 = a; i2 = s; }
        }
        float mx = -3.4e38f;
#pragma unroll
        for (int s = 0; s < T_C; ++s)
            if (s != i1 && s != i2 && sc[s] > mx) mx = sc[s];
        float e[T_C]; float sum = 0.f;
#pragma unroll
        for (int s = 0; s < T_C; ++s) {
            float v = (s != i1 && s != i2) ? __expf(sc[s] - mx) : 0.f;
            e[s] = v; sum += v;
        }
        float inv = 1.f / sum;
        float* ao = attn_out + (((size_t)n * HEADS_C + h) * T_C + t) * T_C;
#pragma unroll
        for (int s = 0; s < T_C; ++s) {
            float v = e[s] * inv;
            ssh[w][h * T_C * T_C + t * T_C + s] = v;
            ao[s] = v;
        }
    }
    __syncthreads();

    // Context: ctx[t][c..c+3] = sum_s attn[h(c)][t][s] * V[s][c..c+3]
    for (int i4 = lane; i4 < T_C * EMBED_C / 4; i4 += 32) {
        int base = i4 * 4;
        int t = base / EMBED_C, c = base % EMBED_C, h = c >> 6;
        v4f acc = {};
#pragma unroll
        for (int s = 0; s < T_C; ++s) {
            float wgt = ssh[w][h * T_C * T_C + t * T_C + s];
            v4f vv = *(const v4f*)&vsh[w][s * EMBED_C + c];
            acc += wgt * vv;
        }
        *(v4f*)&ctx_out[rowbase + base] = acc;
    }
}

// ---------------------------------------------------------------------------
extern "C" void kernel_launch(void* const* d_in, const int* in_sizes, int n_in,
                              void* d_out, int out_size, void* d_ws, size_t ws_size,
                              hipStream_t stream) {
    const float* X  = (const float*)d_in[0];
    const float* ac = (const float*)d_in[1];
    const float* Wk = (const float*)d_in[2];
    const float* bk = (const float*)d_in[3];
    const float* Wq = (const float*)d_in[4];
    const float* bq = (const float*)d_in[5];
    const float* Wg = (const float*)d_in[6];
    const float* bg = (const float*)d_in[7];
    const int* batch = (const int*)d_in[8];

    char*  ws    = (char*)d_ws;
    bf16*  Wc    = (bf16*)ws;                              // 1 MiB
    float* gates = (float*)(ws + (1 << 20));               // 12 KiB
    bf16*  Qbf   = (bf16*)(ws + (1 << 20) + (1 << 16));    // 184.3 MB
    bf16*  Kbf   = Qbf + (size_t)MROWS_C * EMBED_C;        // 184.3 MB

    float* ctx  = (float*)d_out;
    float* attn = ctx + (size_t)MROWS_C * EMBED_C;

    hipLaunchKernelGGL(cvt_weights, dim3((EMBED_C * EMBED_C + 255) / 256), dim3(256),
                       0, stream, Wq, Wk, Wc);
    hipLaunchKernelGGL(gate_kernel, dim3((NGRAPH_C * T_C + 255) / 256), dim3(256),
                       0, stream, ac, Wg, bg, gates);
    hipLaunchKernelGGL(proj_gemm, dim3(MROWS_C / MTILE_C), dim3(512),
                       0, stream, X, Wc, bq, bk, Qbf, Kbf);
    hipLaunchKernelGGL(attn_kernel, dim3(N_NODES_C / 2), dim3(64),
                       0, stream, X, Qbf, Kbf, gates, batch, ctx, attn);
}
